// Model_39676907882717
// MI455X (gfx1250) — compile-verified, matrix operand order checked by reading, not run on hardware
//
#include <hip/hip_runtime.h>
#include <hip/hip_bf16.h>
#include <math.h>

typedef __attribute__((ext_vector_type(16))) __bf16 v16bf;
typedef __attribute__((ext_vector_type(8)))  __bf16 v8bf;
typedef __attribute__((ext_vector_type(8)))  float  v8f;
typedef __attribute__((ext_vector_type(4)))  float  v4f;

#define LOG2E 1.4426950408889634f
#define SEQ   2048
#define DIM   128
#define NBH   32          // B*H
#define NROWS 65536       // B*H*S

__device__ __forceinline__ v8bf cvt8f(const float* __restrict__ p) {
    v4f a = *(const v4f*)p;
    v4f b = *(const v4f*)(p + 4);
    v8bf r;
    r[0]=(__bf16)a[0]; r[1]=(__bf16)a[1]; r[2]=(__bf16)a[2]; r[3]=(__bf16)a[3];
    r[4]=(__bf16)b[0]; r[5]=(__bf16)b[1]; r[6]=(__bf16)b[2]; r[7]=(__bf16)b[3];
    return r;
}

__device__ __forceinline__ v16bf combine16(v8bf lo, v8bf hi) {
    return __builtin_shufflevector(lo, hi, 0,1,2,3,4,5,6,7,8,9,10,11,12,13,14,15);
}

__device__ __forceinline__ v8f wmma_bf16(v16bf a, v16bf b, v8f c) {
    return __builtin_amdgcn_wmma_f32_16x16x32_bf16(false, a, false, b, (short)0, c, false, false);
}

// ---------------------------------------------------------------------------
// GEMM: out[row, e] = (sum_d x[row, d] * W[e, d] + bias[e]) * scale
// Block = 8 waves = 128 rows; W staged once per block as bf16 in LDS.
// ---------------------------------------------------------------------------
template<bool IN_BF16>
__global__ void __launch_bounds__(256, 1)
proj_kernel(const void* __restrict__ xin,
            const float* __restrict__ W,
            const float* __restrict__ bias,
            float scale,
            void* __restrict__ outp,
            int out_is_f32) {
    __shared__ __align__(16) __bf16 ldsW[DIM * DIM];   // 32 KB, [e][d]

    // ---- stage W as bf16 (once per block, shared by all 8 waves) ----
    {
        const int t  = threadIdx.x;
        const int e  = t >> 1;
        const int d0 = (t & 1) * 64;
        const float* wr = W + (size_t)e * DIM + d0;
        __bf16* dst = ldsW + (size_t)e * DIM + d0;
        #pragma unroll
        for (int i = 0; i < 8; ++i)
            *(v8bf*)(dst + i * 8) = cvt8f(wr + i * 8);
    }

    const int lane = threadIdx.x & 31;
    const int wave = threadIdx.x >> 5;
    const int half = lane >> 4;
    const int l16  = lane & 15;
    const int rowBase = (blockIdx.x * 8 + wave) * 16;

    // ---- A operand: 16 rows x 128 (4 k-steps of 32); overlaps W staging ----
    v16bf a[4];
    const int arow = rowBase + l16;
    if constexpr (IN_BF16) {
        const __bf16* x = (const __bf16*)xin + (size_t)arow * DIM;
        #pragma unroll
        for (int kk = 0; kk < 4; ++kk) {
            int d0 = kk * 32 + half * 8;
            a[kk] = combine16(*(const v8bf*)(x + d0), *(const v8bf*)(x + d0 + 16));
        }
    } else {
        const float* x = (const float*)xin + (size_t)arow * DIM;
        #pragma unroll
        for (int kk = 0; kk < 4; ++kk) {
            int d0 = kk * 32 + half * 8;
            a[kk] = combine16(cvt8f(x + d0), cvt8f(x + d0 + 16));
        }
    }
    __syncthreads();

    v8f acc[8];
    #pragma unroll
    for (int n = 0; n < 8; ++n) acc[n] = (v8f)0.0f;

    // ---- B operand: B[K=d][N=e] = W[e][d] from LDS ----
    #pragma unroll
    for (int kk = 0; kk < 4; ++kk) {
        int d0 = kk * 32 + half * 8;
        #pragma unroll
        for (int n = 0; n < 8; ++n) {
            const __bf16* wrow = ldsW + (size_t)(n * 16 + l16) * DIM;
            v16bf b = combine16(*(const v8bf*)(wrow + d0), *(const v8bf*)(wrow + d0 + 16));
            acc[n] = wmma_bf16(a[kk], b, acc[n]);
        }
    }

    // ---- Epilogue: C layout lane L holds (M = r + 8*half, N = l16) ----
    #pragma unroll
    for (int n = 0; n < 8; ++n) {
        const int col = n * 16 + l16;
        const float bv = bias[col];
        #pragma unroll
        for (int r = 0; r < 8; ++r) {
            float v = (acc[n][r] + bv) * scale;
            size_t idx = (size_t)(rowBase + r + half * 8) * DIM + col;
            if (out_is_f32) ((float*)outp)[idx] = v;
            else            ((__bf16*)outp)[idx] = (__bf16)v;
        }
    }
}

// ---------------------------------------------------------------------------
// Flash attention, transposed-score formulation, double-buffered staging.
// Block = 256 threads = 8 waves; each wave owns 16 query rows of one (b,h).
// While block i is computed from lds[cur], block i+1 is loaded from HBM into
// registers and stored to lds[cur^1] after the WMMAs (latency hidden).
// ---------------------------------------------------------------------------
__global__ void __launch_bounds__(256, 1)
flash_kernel(const void* __restrict__ qb_,
             const void* __restrict__ kb_,
             const void* __restrict__ vb_,
             void* __restrict__ ob_) {
    __shared__ __align__(16) __bf16 ldsK[2][32 * DIM];     // 2 x 8 KB, [t][d]
    __shared__ __align__(16) __bf16 ldsVT[2][DIM * 32];    // 2 x 8 KB, [d][t]

    const __bf16* qb = (const __bf16*)qb_;
    const __bf16* kb = (const __bf16*)kb_;
    const __bf16* vb = (const __bf16*)vb_;
    __bf16*       ob = (__bf16*)ob_;

    const int lane = threadIdx.x & 31;
    const int wave = threadIdx.x >> 5;
    const int half = lane >> 4;
    const int l16  = lane & 15;
    const int qblk = blockIdx.x & 15;
    const int bh   = blockIdx.x >> 4;
    const size_t headOff = (size_t)bh * SEQ * DIM;

    // staging assignment: thread covers (row sr, 16 cols at sc) of 32x128 tile
    const int sr = threadIdx.x >> 3;
    const int sc = (threadIdx.x & 7) * 16;
    const __bf16* kst = kb + headOff + (size_t)sr * DIM + sc;
    const __bf16* vst = vb + headOff + (size_t)sr * DIM + sc;

    // ---- prologue: stage key block 0 into buffer 0 ----
    v8bf kr0 = *(const v8bf*)kst,      kr1 = *(const v8bf*)(kst + 8);
    v8bf vr0 = *(const v8bf*)vst,      vr1 = *(const v8bf*)(vst + 8);
    *(v8bf*)(&ldsK[0][sr * DIM + sc])     = kr0;
    *(v8bf*)(&ldsK[0][sr * DIM + sc + 8]) = kr1;
    #pragma unroll
    for (int i = 0; i < 8; ++i) {
        ldsVT[0][(sc + i) * 32 + sr]     = vr0[i];
        ldsVT[0][(sc + 8 + i) * 32 + sr] = vr1[i];
    }

    // ---- Q as B operand: B[K=d][N=s], lane owns query row s = l16 ----
    v16bf bq[4];
    {
        const __bf16* q = qb + headOff + (size_t)(qblk * 128 + wave * 16 + l16) * DIM;
        #pragma unroll
        for (int kk = 0; kk < 4; ++kk) {
            int d0 = kk * 32 + half * 8;
            bq[kk] = combine16(*(const v8bf*)(q + d0), *(const v8bf*)(q + d0 + 16));
        }
    }

    // ones matrix (B operand) for row-sum accumulation
    v16bf ones;
    #pragma unroll
    for (int i = 0; i < 16; ++i) ones[i] = (__bf16)1.0f;

    v8f o[8];
    #pragma unroll
    for (int n = 0; n < 8; ++n) o[n] = (v8f)0.0f;
    v8f o_l = (v8f)0.0f;          // running row-sum tile (C layout, rows = s)
    float m = -INFINITY;          // running max for row s = l16

    const int NB = SEQ / 32;
    for (int ib = 0; ib < NB; ++ib) {
        const int cur = ib & 1;
        __syncthreads();   // lds[cur] published; lds[cur^1] free to overwrite

        // ---- issue global loads for block ib+1 (consumed after compute) ----
        if (ib + 1 < NB) {
            const __bf16* kn = kst + (size_t)(ib + 1) * 32 * DIM;
            const __bf16* vn = vst + (size_t)(ib + 1) * 32 * DIM;
            kr0 = *(const v8bf*)kn;       kr1 = *(const v8bf*)(kn + 8);
            vr0 = *(const v8bf*)vn;       vr1 = *(const v8bf*)(vn + 8);
            if (ib + 2 < NB) {
                __builtin_prefetch(kst + (size_t)(ib + 2) * 32 * DIM, 0, 1);
                __builtin_prefetch(vst + (size_t)(ib + 2) * 32 * DIM, 0, 1);
            }
        }

        // ---- transposed scores: S^T[t][s], K tiles as A, Q as B ----
        v8f s0 = (v8f)0.0f, s1 = (v8f)0.0f;
        #pragma unroll
        for (int kk = 0; kk < 4; ++kk) {
            int d0 = kk * 32 + half * 8;
            const __bf16* k0 = &ldsK[cur][(size_t)l16 * DIM];
            const __bf16* k1 = &ldsK[cur][(size_t)(16 + l16) * DIM];
            v16bf a0 = combine16(*(const v8bf*)(k0 + d0), *(const v8bf*)(k0 + d0 + 16));
            v16bf a1 = combine16(*(const v8bf*)(k1 + d0), *(const v8bf*)(k1 + d0 + 16));
            s0 = wmma_bf16(a0, bq[kk], s0);
            s1 = wmma_bf16(a1, bq[kk], s1);
        }
        // lane (half,l16) element r: s0 = score(t=r+8*half, s=l16),
        //                            s1 = score(t=16+r+8*half, s=l16)

        // ---- row max: in-register tree + one cross-half exchange ----
        float mx = fmaxf(s0[0], s1[0]);
        #pragma unroll
        for (int r = 1; r < 8; ++r) mx = fmaxf(mx, fmaxf(s0[r], s1[r]));
        mx = fmaxf(mx, __shfl_xor(mx, 16, 32));
        float newm = fmaxf(m, mx);
        float corr = exp2f((m - newm) * LOG2E);
        m = newm;

        // ---- probabilities, packed directly into A-operand layout ----
        v16bf ap;
        #pragma unroll
        for (int r = 0; r < 8; ++r) {
            ap[r]     = (__bf16)exp2f((s0[r] - newm) * LOG2E);
            ap[r + 8] = (__bf16)exp2f((s1[r] - newm) * LOG2E);
        }

        // ---- broadcast corr (per-lane, row=l16) into C-layout rows ----
        float corrM[8];
        #pragma unroll
        for (int r = 0; r < 8; ++r)
            corrM[r] = __shfl(corr, 24 * half + r, 32);

        #pragma unroll
        for (int n = 0; n < 8; ++n)
            #pragma unroll
            for (int r = 0; r < 8; ++r) o[n][r] *= corrM[r];
        #pragma unroll
        for (int r = 0; r < 8; ++r) o_l[r] *= corrM[r];

        // ---- row-sum tile and P @ V ----
        o_l = wmma_bf16(ap, ones, o_l);
        #pragma unroll
        for (int nd = 0; nd < 8; ++nd) {
            const __bf16* vt = &ldsVT[cur][(size_t)(nd * 16 + l16) * 32];
            int t0 = half * 8;
            v16bf bv = combine16(*(const v8bf*)(vt + t0), *(const v8bf*)(vt + t0 + 16));
            o[nd] = wmma_bf16(ap, bv, o[nd]);
        }

        // ---- store staged block ib+1 into the other buffer ----
        if (ib + 1 < NB) {
            const int nxt = cur ^ 1;
            *(v8bf*)(&ldsK[nxt][sr * DIM + sc])     = kr0;
            *(v8bf*)(&ldsK[nxt][sr * DIM + sc + 8]) = kr1;
            #pragma unroll
            for (int i = 0; i < 8; ++i) {
                ldsVT[nxt][(sc + i) * 32 + sr]     = vr0[i];
                ldsVT[nxt][(sc + 8 + i) * 32 + sr] = vr1[i];
            }
        }
    }

    // ---- normalize (l lives in o_l, same C layout as o) and store ----
    #pragma unroll
    for (int r = 0; r < 8; ++r) {
        float inv = 1.0f / o_l[r];
        __bf16* orow = ob + headOff +
                       (size_t)(qblk * 128 + wave * 16 + r + half * 8) * DIM;
        #pragma unroll
        for (int nd = 0; nd < 8; ++nd)
            orow[nd * 16 + l16] = (__bf16)(o[nd][r] * inv);
    }
}

// ---------------------------------------------------------------------------
extern "C" void kernel_launch(void* const* d_in, const int* in_sizes, int n_in,
                              void* d_out, int out_size, void* d_ws, size_t ws_size,
                              hipStream_t stream) {
    const float* query = (const float*)d_in[0];
    const float* key_  = (const float*)d_in[1];
    const float* value = (const float*)d_in[2];
    const float* Wq = (const float*)d_in[3];
    const float* bq = (const float*)d_in[4];
    const float* Wk = (const float*)d_in[5];
    const float* bk = (const float*)d_in[6];
    const float* Wv = (const float*)d_in[7];
    const float* bv = (const float*)d_in[8];
    const float* Wo = (const float*)d_in[9];
    const float* bo = (const float*)d_in[10];

    const size_t N = (size_t)NROWS * DIM;   // 8,388,608 elements
    char* ws = (char*)d_ws;
    void* qbuf = ws;                        // bf16 Q
    void* kbuf = ws + N * 2;                // bf16 K
    void* vbuf = ws + N * 4;                // bf16 V
    void* obuf = ws + N * 6;                // bf16 attention output

    const dim3 blk(256);
    const dim3 gproj(NROWS / 16 / 8);       // 512 blocks
    const dim3 gattn(NBH * (SEQ / 128));    // 512 blocks
    const float qscale = 1.0f / sqrtf((float)DIM);

    proj_kernel<false><<<gproj, blk, 0, stream>>>(query, Wq, bq, qscale, qbuf, 0);
    proj_kernel<false><<<gproj, blk, 0, stream>>>(key_,  Wk, bk, 1.0f,   kbuf, 0);
    proj_kernel<false><<<gproj, blk, 0, stream>>>(value, Wv, bv, 1.0f,   vbuf, 0);
    flash_kernel<<<gattn, blk, 0, stream>>>(qbuf, kbuf, vbuf, obuf);
    proj_kernel<true><<<gproj, blk, 0, stream>>>(obuf, Wo, bo, 1.0f, d_out, 1);
}
